// GLGRU_59261958750754
// MI455X (gfx1250) — matrix operand order repaired and checked by reading, not compiled, not run on hardware
//
#include <hip/hip_runtime.h>
#include <hip/hip_bf16.h>

// ---------------------------------------------------------------------------
// GLGRU on MI455X (gfx1250)
//   - persistent kernel, whole T=512 recurrence on-device
//   - bf16 V_WMMA_F32_16X16X32_BF16, fp32 accumulate, 2x2 tile register block
//   - double-buffered k-loop: next k-step's fragments loaded before WMMAs
//   - 4 device-wide barriers per step via L2 atomics (sense-reversing)
//   - weights (7.5MB bf16) stay hot in 192MB L2; per-step GEMMs are L2-fed
// ---------------------------------------------------------------------------

#define TT  512   // time steps
#define BB  256   // batch
#define HH  512   // hidden
#define INN 512   // input

typedef __attribute__((ext_vector_type(16))) __bf16        v16bf;
typedef __attribute__((ext_vector_type(8)))  float         v8f;
typedef __attribute__((ext_vector_type(4)))  unsigned int  v4u;
typedef __attribute__((ext_vector_type(8)))  unsigned int  v8u;

__device__ __forceinline__ unsigned short f32_to_bf16_rne(float f) {
  unsigned int u = __builtin_bit_cast(unsigned int, f);
  unsigned int lsb = (u >> 16) & 1u;
  u += 0x7fffu + lsb;
  return (unsigned short)(u >> 16);
}

__device__ __forceinline__ float sigm(float x) {
  return 1.0f / (1.0f + __expf(-x));
}

// ---------------------------------------------------------------------------
// fp32 -> bf16 conversion (grid-stride)
// ---------------------------------------------------------------------------
__global__ void cvt_f32_to_bf16(const float* __restrict__ src,
                                unsigned short* __restrict__ dst, long n) {
  long i = (long)blockIdx.x * blockDim.x + threadIdx.x;
  long stride = (long)gridDim.x * blockDim.x;
  for (; i < n; i += stride) dst[i] = f32_to_bf16_rne(src[i]);
}

__global__ void init_barrier_k(unsigned int* bar) {
  if (threadIdx.x < 2) bar[threadIdx.x] = 0u;
}

// ---------------------------------------------------------------------------
// device-wide sense-reversing barrier (all workgroups co-resident)
// ---------------------------------------------------------------------------
__device__ __forceinline__ void grid_sync(unsigned int* cnt, unsigned int* gen,
                                          unsigned int nwg) {
  __syncthreads();
  __threadfence();                         // make phase stores device-visible
  if (threadIdx.x == 0) {
    unsigned int g = __hip_atomic_load(gen, __ATOMIC_RELAXED,
                                       __HIP_MEMORY_SCOPE_AGENT);
    unsigned int a = __hip_atomic_fetch_add(cnt, 1u, __ATOMIC_ACQ_REL,
                                            __HIP_MEMORY_SCOPE_AGENT);
    if (a == nwg - 1u) {
      __hip_atomic_store(cnt, 0u, __ATOMIC_RELAXED, __HIP_MEMORY_SCOPE_AGENT);
      __hip_atomic_store(gen, g + 1u, __ATOMIC_RELEASE,
                         __HIP_MEMORY_SCOPE_AGENT);
    } else {
      while (__hip_atomic_load(gen, __ATOMIC_ACQUIRE,
                               __HIP_MEMORY_SCOPE_AGENT) == g)
        __builtin_amdgcn_s_sleep(2);
    }
  }
  __syncthreads();
}

// ---------------------------------------------------------------------------
// WMMA fragment loads per CDNA5 ISA layouts (16-bit A 16x32 / B 32x16)
// ---------------------------------------------------------------------------
__device__ __forceinline__ v16bf load_frag_A(const unsigned short* rowptr,
                                             int k0) {
  // rowptr already includes row m and the lane-half K offset (0 or 8)
  const unsigned short* p = rowptr + k0;
  v4u lo = *(const v4u*)p;        // K = k..k+7   -> VGPR 0..3
  v4u hi = *(const v4u*)(p + 16); // K = k+16..23 -> VGPR 4..7
  v8u r;
  r[0] = lo[0]; r[1] = lo[1]; r[2] = lo[2]; r[3] = lo[3];
  r[4] = hi[0]; r[5] = hi[1]; r[6] = hi[2]; r[7] = hi[3];
  return __builtin_bit_cast(v16bf, r);
}

__device__ __forceinline__ v16bf load_frag_B(const unsigned short* colptr,
                                             int k0) {
  // colptr already includes column n and the lane-half K offset (0 or 8);
  // K = k..k+7 and k+16..k+23 per lane (same order as A).
  const unsigned short* p = colptr + k0;
  v4u lo = *(const v4u*)p;
  v4u hi = *(const v4u*)(p + 16);
  v8u r;
  r[0] = lo[0]; r[1] = lo[1]; r[2] = lo[2]; r[3] = lo[3];
  r[4] = hi[0]; r[5] = hi[1]; r[6] = hi[2]; r[7] = hi[3];
  return __builtin_bit_cast(v16bf, r);
}

#define WMMA_BF16(a, b, c)                                                    \
  __builtin_amdgcn_wmma_f32_16x16x32_bf16(false, (a), false, (b), (short)0,   \
                                          (c), false, false)

__device__ __forceinline__ void store_tile(float* __restrict__ Y, int ldY,
                                           int m0, int n0, int lane, v8f acc,
                                           float bv) {
  float* yp = Y + (size_t)(m0 + ((lane >> 4) << 3)) * ldY + (n0 + (lane & 15));
#pragma unroll
  for (int r = 0; r < 8; ++r) yp[(size_t)r * ldY] = acc[r] + bv;
}

// 2x2 block of 16x16 tiles: D(32x32) = A(32x512) * B(512x32); 2 loads/WMMA.
// Software-pipelined: fragments for k+32 are in flight while k computes.
__device__ __forceinline__ void gemm_block_2x2(
    const unsigned short* __restrict__ A, const unsigned short* __restrict__ W,
    int m0, int n0, int lane, v8f* acc00, v8f* acc01, v8f* acc10, v8f* acc11) {
  const unsigned short* a0 =
      A + (size_t)(m0 + (lane & 15)) * 512 + ((lane >> 4) << 3);
  const unsigned short* a1 = a0 + 16 * 512;
  const unsigned short* b0 =
      W + (size_t)(n0 + (lane & 15)) * 512 + ((lane >> 4) << 3);
  const unsigned short* b1 = b0 + 16 * 512;
  __builtin_prefetch(b0 + 256, 0, 3);  // global_prefetch_b8: far half of cols
  __builtin_prefetch(b1 + 256, 0, 3);
  v16bf A0 = load_frag_A(a0, 0);
  v16bf A1 = load_frag_A(a1, 0);
  v16bf B0 = load_frag_B(b0, 0);
  v16bf B1 = load_frag_B(b1, 0);
  v8f c00 = {}, c01 = {}, c10 = {}, c11 = {};
#pragma unroll
  for (int k0 = 0; k0 < 512 - 32; k0 += 32) {
    v16bf nA0 = load_frag_A(a0, k0 + 32);
    v16bf nA1 = load_frag_A(a1, k0 + 32);
    v16bf nB0 = load_frag_B(b0, k0 + 32);
    v16bf nB1 = load_frag_B(b1, k0 + 32);
    c00 = WMMA_BF16(A0, B0, c00);
    c01 = WMMA_BF16(A0, B1, c01);
    c10 = WMMA_BF16(A1, B0, c10);
    c11 = WMMA_BF16(A1, B1, c11);
    A0 = nA0; A1 = nA1; B0 = nB0; B1 = nB1;
  }
  c00 = WMMA_BF16(A0, B0, c00);
  c01 = WMMA_BF16(A0, B1, c01);
  c10 = WMMA_BF16(A1, B0, c10);
  c11 = WMMA_BF16(A1, B1, c11);
  *acc00 = c00; *acc01 = c01; *acc10 = c10; *acc11 = c11;
}

// 1x2 block (16x32 output), used for the smaller GEMM2. Same pipelining.
__device__ __forceinline__ void gemm_block_1x2(
    const unsigned short* __restrict__ A, const unsigned short* __restrict__ W,
    int m0, int n0, int lane, v8f* acc0, v8f* acc1) {
  const unsigned short* a0 =
      A + (size_t)(m0 + (lane & 15)) * 512 + ((lane >> 4) << 3);
  const unsigned short* b0 =
      W + (size_t)(n0 + (lane & 15)) * 512 + ((lane >> 4) << 3);
  const unsigned short* b1 = b0 + 16 * 512;
  __builtin_prefetch(b0 + 256, 0, 3);
  v16bf A0 = load_frag_A(a0, 0);
  v16bf B0 = load_frag_B(b0, 0);
  v16bf B1 = load_frag_B(b1, 0);
  v8f c0 = {}, c1 = {};
#pragma unroll
  for (int k0 = 0; k0 < 512 - 32; k0 += 32) {
    v16bf nA0 = load_frag_A(a0, k0 + 32);
    v16bf nB0 = load_frag_B(b0, k0 + 32);
    v16bf nB1 = load_frag_B(b1, k0 + 32);
    c0 = WMMA_BF16(A0, B0, c0);
    c1 = WMMA_BF16(A0, B1, c1);
    A0 = nA0; B0 = nB0; B1 = nB1;
  }
  c0 = WMMA_BF16(A0, B0, c0);
  c1 = WMMA_BF16(A0, B1, c1);
  *acc0 = c0; *acc1 = c1;
}

// ---------------------------------------------------------------------------
// Persistent recurrent kernel
//   Y1 layout per batch row (6144 cols): [0,2048) x@Wx.T   (z,r,h,u)
//                                        [2048,3584) h@Wh.T  (z,r,u)
//                                        [3584,5120) g@Wgh.T (z,r,u)
//                                        [5120,6144) g@Wg.T  (z2,r2)
//   Y2 layout (1536 cols): hd@Whd.T (z,r,g)
// ---------------------------------------------------------------------------
__global__ void __launch_bounds__(512)
glgru_persistent(const unsigned short* __restrict__ x_bf,   // [T][B][IN]
                 const unsigned short* __restrict__ Wx_bf,  // [2048][512]
                 const unsigned short* __restrict__ Wh_bf,  // [1536][512]
                 const unsigned short* __restrict__ Wgh_bf, // [1536][512]
                 const unsigned short* __restrict__ Whd_bf, // [1536][512]
                 const unsigned short* __restrict__ Wg_bf,  // [1024][512]
                 const float* __restrict__ bx, const float* __restrict__ bh,
                 const float* __restrict__ bgh, const float* __restrict__ bhd,
                 const float* __restrict__ bg,
                 unsigned short* __restrict__ h_bf,
                 unsigned short* __restrict__ g_bf,
                 unsigned short* __restrict__ hd_bf,
                 float* __restrict__ h_f, float* __restrict__ g_f,
                 float* __restrict__ Y1, float* __restrict__ Y2,
                 float* __restrict__ out_hs, float* __restrict__ out_h,
                 float* __restrict__ out_g, unsigned int* __restrict__ barrier) {
  const int lane = threadIdx.x & 31;
  const unsigned int nwg = gridDim.x;
  const int wave = (int)((blockIdx.x * blockDim.x + threadIdx.x) >> 5);
  const int nwaves = (int)((gridDim.x * blockDim.x) >> 5);
  const int tid = (int)(blockIdx.x * blockDim.x + threadIdx.x);
  const int nthr = (int)(gridDim.x * blockDim.x);
  unsigned int* cnt = barrier;
  unsigned int* gen = barrier + 1;

  for (int t = 0; t < TT; ++t) {
    const unsigned short* xt = x_bf + (size_t)t * BB * INN;

    // ---- Phase A: fused GEMM1, 8 x 192 blocks of 32x32, K=512 -------------
    // n-blocks (32 cols) never straddle a source region (boundaries at
    // n-tile 128/224/320 are even).
    for (int blk = wave; blk < 8 * 192; blk += nwaves) {
      int mb = blk / 192;          // 32-row block
      int nb = blk % 192;          // 32-col block
      const unsigned short* Asrc;
      const unsigned short* Wsrc;
      const float* bsrc;
      int ncol;                    // local column within this source GEMM
      if (nb < 64)       { Asrc = xt;   Wsrc = Wx_bf;  bsrc = bx;  ncol = nb * 32; }
      else if (nb < 112) { Asrc = h_bf; Wsrc = Wh_bf;  bsrc = bh;  ncol = (nb - 64) * 32; }
      else if (nb < 160) { Asrc = g_bf; Wsrc = Wgh_bf; bsrc = bgh; ncol = (nb - 112) * 32; }
      else               { Asrc = g_bf; Wsrc = Wg_bf;  bsrc = bg;  ncol = (nb - 160) * 32; }
      v8f c00, c01, c10, c11;
      gemm_block_2x2(Asrc, Wsrc, mb * 32, ncol, lane, &c00, &c01, &c10, &c11);
      float bv0 = bsrc[ncol + (lane & 15)];
      float bv1 = bsrc[ncol + 16 + (lane & 15)];
      int yn = nb * 32;
      store_tile(Y1, 6144, mb * 32,      yn,      lane, c00, bv0);
      store_tile(Y1, 6144, mb * 32,      yn + 16, lane, c01, bv1);
      store_tile(Y1, 6144, mb * 32 + 16, yn,      lane, c10, bv0);
      store_tile(Y1, 6144, mb * 32 + 16, yn + 16, lane, c11, bv1);
    }
    grid_sync(cnt, gen, nwg);

    // ---- Phase B: gates, h_new, h_d (vectorized x4) -----------------------
    for (int v = tid; v < (BB * HH) / 4; v += nthr) {
      int i = v * 4;
      int b = i >> 9, j = i & 511;
      const float* y = Y1 + (size_t)b * 6144 + j;
      float4 xz = *(const float4*)(y);
      float4 xr = *(const float4*)(y + 512);
      float4 xh = *(const float4*)(y + 1024);
      float4 xu = *(const float4*)(y + 1536);
      float4 hz = *(const float4*)(y + 2048);
      float4 hr = *(const float4*)(y + 2560);
      float4 hu = *(const float4*)(y + 3072);
      float4 gz = *(const float4*)(y + 3584);
      float4 gr = *(const float4*)(y + 4096);
      float4 gu = *(const float4*)(y + 4608);
      float4 ho = *(const float4*)(h_f + i);
      float4 go = *(const float4*)(g_f + i);
      float4 hn, hd;
      {
        float z0 = sigm(xz.x + hz.x + gz.x), r0 = sigm(xr.x + hr.x + gr.x), u0 = sigm(xu.x + hu.x + gu.x);
        float z1 = sigm(xz.y + hz.y + gz.y), r1 = sigm(xr.y + hr.y + gr.y), u1 = sigm(xu.y + hu.y + gu.y);
        float z2 = sigm(xz.z + hz.z + gz.z), r2 = sigm(xr.z + hr.z + gr.z), u2 = sigm(xu.z + hu.z + gu.z);
        float z3 = sigm(xz.w + hz.w + gz.w), r3 = sigm(xr.w + hr.w + gr.w), u3 = sigm(xu.w + hu.w + gu.w);
        hn.x = (1.0f - z0) * ho.x + z0 * tanhf(xh.x + r0 * ho.x + u0 * go.x);
        hn.y = (1.0f - z1) * ho.y + z1 * tanhf(xh.y + r1 * ho.y + u1 * go.y);
        hn.z = (1.0f - z2) * ho.z + z2 * tanhf(xh.z + r2 * ho.z + u2 * go.z);
        hn.w = (1.0f - z3) * ho.w + z3 * tanhf(xh.w + r3 * ho.w + u3 * go.w);
      }
      hd.x = hn.x - ho.x; hd.y = hn.y - ho.y; hd.z = hn.z - ho.z; hd.w = hn.w - ho.w;
      *(float4*)(h_f + i) = hn;
      *(float4*)(out_hs + (size_t)t * (BB * HH) + i) = hn;
      ushort4 hnb = make_ushort4(f32_to_bf16_rne(hn.x), f32_to_bf16_rne(hn.y),
                                 f32_to_bf16_rne(hn.z), f32_to_bf16_rne(hn.w));
      ushort4 hdb = make_ushort4(f32_to_bf16_rne(hd.x), f32_to_bf16_rne(hd.y),
                                 f32_to_bf16_rne(hd.z), f32_to_bf16_rne(hd.w));
      *(ushort4*)(h_bf + i) = hnb;
      *(ushort4*)(hd_bf + i) = hdb;
    }
    grid_sync(cnt, gen, nwg);

    // ---- Phase C: GEMM2, hd @ Whd.T, 16 x 48 blocks of 16x32 --------------
    for (int blk = wave; blk < 16 * 48; blk += nwaves) {
      int mt = blk / 48;
      int nb = blk % 48;
      v8f c0, c1;
      gemm_block_1x2(hd_bf, Whd_bf, mt * 16, nb * 32, lane, &c0, &c1);
      float bv0 = bhd[nb * 32 + (lane & 15)];
      float bv1 = bhd[nb * 32 + 16 + (lane & 15)];
      store_tile(Y2, 1536, mt * 16, nb * 32,      lane, c0, bv0);
      store_tile(Y2, 1536, mt * 16, nb * 32 + 16, lane, c1, bv1);
    }
    grid_sync(cnt, gen, nwg);

    // ---- Phase D: g_new (vectorized x4) -----------------------------------
    for (int v = tid; v < (BB * HH) / 4; v += nthr) {
      int i = v * 4;
      int b = i >> 9, j = i & 511;
      const float* y2 = Y2 + (size_t)b * 1536 + j;
      const float* y1 = Y1 + (size_t)b * 6144 + 5120 + j;
      float4 hdz = *(const float4*)(y2);
      float4 hdr = *(const float4*)(y2 + 512);
      float4 hdg = *(const float4*)(y2 + 1024);
      float4 gz2 = *(const float4*)(y1);
      float4 gr2 = *(const float4*)(y1 + 512);
      float4 go = *(const float4*)(g_f + i);
      float4 gn;
      {
        float z0 = sigm(hdz.x + gz2.x), r0 = sigm(hdr.x + gr2.x);
        float z1 = sigm(hdz.y + gz2.y), r1 = sigm(hdr.y + gr2.y);
        float z2 = sigm(hdz.z + gz2.z), r2 = sigm(hdr.z + gr2.z);
        float z3 = sigm(hdz.w + gz2.w), r3 = sigm(hdr.w + gr2.w);
        gn.x = (1.0f - z0) * go.x + z0 * tanhf(hdg.x + r0 * go.x);
        gn.y = (1.0f - z1) * go.y + z1 * tanhf(hdg.y + r1 * go.y);
        gn.z = (1.0f - z2) * go.z + z2 * tanhf(hdg.z + r2 * go.z);
        gn.w = (1.0f - z3) * go.w + z3 * tanhf(hdg.w + r3 * go.w);
      }
      *(float4*)(g_f + i) = gn;
      ushort4 gnb = make_ushort4(f32_to_bf16_rne(gn.x), f32_to_bf16_rne(gn.y),
                                 f32_to_bf16_rne(gn.z), f32_to_bf16_rne(gn.w));
      *(ushort4*)(g_bf + i) = gnb;
    }
    grid_sync(cnt, gen, nwg);
  }

  // final h, g outputs (vectorized)
  for (int v = tid; v < (BB * HH) / 4; v += nthr) {
    int i = v * 4;
    *(float4*)(out_h + i) = *(const float4*)(h_f + i);
    *(float4*)(out_g + i) = *(const float4*)(g_f + i);
  }
}

// ---------------------------------------------------------------------------
// Host-side launch
// ---------------------------------------------------------------------------
extern "C" void kernel_launch(void* const* d_in, const int* in_sizes, int n_in,
                              void* d_out, int out_size, void* d_ws,
                              size_t ws_size, hipStream_t stream) {
  (void)in_sizes; (void)n_in; (void)out_size; (void)ws_size;

  const float* x   = (const float*)d_in[0];
  const float* h0  = (const float*)d_in[1];
  const float* g0  = (const float*)d_in[2];
  const float* Wx  = (const float*)d_in[3];
  const float* bx  = (const float*)d_in[4];
  const float* Wh  = (const float*)d_in[5];
  const float* bh  = (const float*)d_in[6];
  const float* Wgh = (const float*)d_in[7];
  const float* bgh = (const float*)d_in[8];
  const float* Whd = (const float*)d_in[9];
  const float* bhd = (const float*)d_in[10];
  const float* Wg  = (const float*)d_in[11];
  const float* bg  = (const float*)d_in[12];

  char* w = (char*)d_ws;
  size_t off = 0;
  auto alloc = [&](size_t bytes) {
    void* p = w + off;
    off = (off + bytes + 255) & ~(size_t)255;
    return p;
  };

  unsigned int*   bar    = (unsigned int*)alloc(256);
  unsigned short* x_bf   = (unsigned short*)alloc((size_t)TT * BB * INN * 2);
  unsigned short* Wx_bf  = (unsigned short*)alloc((size_t)2048 * 512 * 2);
  unsigned short* Wh_bf  = (unsigned short*)alloc((size_t)1536 * 512 * 2);
  unsigned short* Wgh_bf = (unsigned short*)alloc((size_t)1536 * 512 * 2);
  unsigned short* Whd_bf = (unsigned short*)alloc((size_t)1536 * 512 * 2);
  unsigned short* Wg_bf  = (unsigned short*)alloc((size_t)1024 * 512 * 2);
  unsigned short* h_bf   = (unsigned short*)alloc((size_t)BB * HH * 2);
  unsigned short* g_bf   = (unsigned short*)alloc((size_t)BB * HH * 2);
  unsigned short* hd_bf  = (unsigned short*)alloc((size_t)BB * HH * 2);
  float*          h_f    = (float*)alloc((size_t)BB * HH * 4);
  float*          g_f    = (float*)alloc((size_t)BB * HH * 4);
  float*          Y1     = (float*)alloc((size_t)BB * 6144 * 4);
  float*          Y2     = (float*)alloc((size_t)BB * 1536 * 4);

  float* out_hs = (float*)d_out;
  float* out_h  = out_hs + (size_t)TT * BB * HH;
  float* out_g  = out_h + (size_t)BB * HH;

  init_barrier_k<<<1, 64, 0, stream>>>(bar);

  // bf16 conversions (x is the big one: 67M elements)
  cvt_f32_to_bf16<<<8192, 256, 0, stream>>>(x, x_bf, (long)TT * BB * INN);
  cvt_f32_to_bf16<<<1024, 256, 0, stream>>>(Wx, Wx_bf, 2048L * 512);
  cvt_f32_to_bf16<<<1024, 256, 0, stream>>>(Wh, Wh_bf, 1536L * 512);
  cvt_f32_to_bf16<<<1024, 256, 0, stream>>>(Wgh, Wgh_bf, 1536L * 512);
  cvt_f32_to_bf16<<<1024, 256, 0, stream>>>(Whd, Whd_bf, 1536L * 512);
  cvt_f32_to_bf16<<<1024, 256, 0, stream>>>(Wg, Wg_bf, 1024L * 512);
  cvt_f32_to_bf16<<<256, 256, 0, stream>>>(h0, h_bf, (long)BB * HH);
  cvt_f32_to_bf16<<<256, 256, 0, stream>>>(g0, g_bf, (long)BB * HH);
  hipMemcpyAsync(h_f, h0, (size_t)BB * HH * 4, hipMemcpyDeviceToDevice, stream);
  hipMemcpyAsync(g_f, g0, (size_t)BB * HH * 4, hipMemcpyDeviceToDevice, stream);

  // persistent kernel: 128 workgroups x 512 threads (16 wave32 each, 4 waves
  // per SIMD32 for latency hiding), assumed co-resident for the grid barrier.
  glgru_persistent<<<128, 512, 0, stream>>>(
      x_bf, Wx_bf, Wh_bf, Wgh_bf, Whd_bf, Wg_bf, bx, bh, bgh, bhd, bg, h_bf,
      g_bf, hd_bf, h_f, g_f, Y1, Y2, out_hs, out_h, out_g, bar);
}
